// WavelengthConditionedPatchEmbed_38766374814157
// MI455X (gfx1250) — matrix operand order repaired and checked by reading, not compiled, not run on hardware
//
#include <hip/hip_runtime.h>

// Problem constants
#define IMG   224
#define PATCH 16
#define CCH   12
#define EMB   768
#define HIDN  128
#define NHH   14
#define NTOK  196          // patches per image
#define BATCH 32
#define KDIM  3072         // C*P*P
#define W2N   196608       // P*P*E

#if defined(__has_builtin)
#if __has_builtin(__builtin_amdgcn_global_load_async_to_lds_b128) && \
    __has_builtin(__builtin_amdgcn_s_wait_asynccnt)
#define USE_ASYNC_LDS 1
#endif
#endif
#ifndef USE_ASYNC_LDS
#define USE_ASYNC_LDS 0
#endif

typedef __attribute__((ext_vector_type(16))) __bf16 v16bf;
typedef __attribute__((ext_vector_type(2)))  __bf16 v2bf;
typedef __attribute__((ext_vector_type(8)))  float  v8f;
typedef int v4i __attribute__((vector_size(16)));    // matches builtin param type

// Native conversions -> v_cvt_pk_bf16_f32 / v_cvt_bf16_f32 (RTNE in HW)
static __device__ __forceinline__ unsigned int pk2bf(float lo, float hi) {
    v2bf v;
    v[0] = (__bf16)lo;
    v[1] = (__bf16)hi;
    return __builtin_bit_cast(unsigned int, v);
}
static __device__ __forceinline__ unsigned short f2bf(float f) {
    __bf16 h = (__bf16)f;
    return __builtin_bit_cast(unsigned short, h);
}

#if USE_ASYNC_LDS
// Generic LDS pointers keep the LDS byte-offset in the low 32 bits, so the
// int round-trip yields a valid addrspace(3) pointer (same trick CK uses).
static __device__ __forceinline__ __attribute__((address_space(3))) v4i*
to_lds(void* p) {
    return (__attribute__((address_space(3))) v4i*)(unsigned int)(uintptr_t)p;
}
static __device__ __forceinline__ __attribute__((address_space(1))) v4i*
to_glb(const void* p) {
    return (__attribute__((address_space(1))) v4i*)(uintptr_t)p;
}
#endif

// ---------------------------------------------------------------------------
// K1: h = gelu(wl/2500 * w1 + b1)   -> hbuf[12][128] f32
// ---------------------------------------------------------------------------
__global__ void k_wlmlp(const float* __restrict__ wl, const float* __restrict__ w1,
                        const float* __restrict__ b1, float* __restrict__ hbuf) {
    int hid = threadIdx.x;                    // 128 threads
    float w = w1[hid], bb = b1[hid];
    for (int c = 0; c < CCH; ++c) {
        float z = (wl[c] * (1.0f / 2500.0f)) * w + bb;
        float g = 0.5f * z * (1.0f + erff(z * 0.70710678118654752f));
        hbuf[c * HIDN + hid] = g;
    }
}

// ---------------------------------------------------------------------------
// K2: kernels = h @ w2 + b2, stored TRANSPOSED as bf16: Bt[e][c*256+p*16+q]
// ---------------------------------------------------------------------------
__global__ __launch_bounds__(256) void k_kernels(const float* __restrict__ hbuf,
                                                 const float* __restrict__ w2,
                                                 const float* __restrict__ b2,
                                                 unsigned short* __restrict__ Bt) {
    int idx = blockIdx.x * 256 + threadIdx.x;       // 0..196607 = e*256 + pq
    float bias = b2[idx];
    float acc[CCH];
#pragma unroll
    for (int c = 0; c < CCH; ++c) acc[c] = bias;
    for (int hid = 0; hid < HIDN; ++hid) {
        float w = w2[(size_t)hid * W2N + idx];      // coalesced vector load
#pragma unroll
        for (int c = 0; c < CCH; ++c)
            acc[c] = fmaf(hbuf[c * HIDN + hid], w, acc[c]);  // uniform -> SMEM
    }
    int e = idx >> 8, pq = idx & 255;
#pragma unroll
    for (int c = 0; c < CCH; ++c)
        Bt[(size_t)e * KDIM + c * 256 + pq] = f2bf(acc[c]);
}

// ---------------------------------------------------------------------------
// K3: cls rows: out[b][0][e] = cls_token[e] + pos_embed[0][e]
// ---------------------------------------------------------------------------
__global__ void k_cls(const float* __restrict__ cls_token,
                      const float* __restrict__ pos_embed, float* __restrict__ out) {
    int idx = blockIdx.x * 256 + threadIdx.x;       // 0..24575
    int b = idx / EMB, e = idx % EMB;
    out[(size_t)b * 197 * EMB + e] = cls_token[e] + pos_embed[e];
}

// ---------------------------------------------------------------------------
// K4: GEMM  A[6272][3072] (patches of x, f32->bf16)  x  Bt[3072][768]
//     128x128 tile/WG, 8 waves, double-buffered LDS, async B staging.
// ---------------------------------------------------------------------------
__global__ __launch_bounds__(256) void k_gemm(const float* __restrict__ x,
                                              const unsigned short* __restrict__ Bt,
                                              const float* __restrict__ proj_bias,
                                              const float* __restrict__ pos_embed,
                                              float* __restrict__ out) {
    __shared__ alignas(16) unsigned short sA[2][128 * 32];   // [m][k]  2 x 8 KB
    __shared__ alignas(16) unsigned short sB[2][128 * 32];   // [n][k]  2 x 8 KB

    const int t    = threadIdx.x;
    const int lane = t & 31;
    const int wave = t >> 5;
    const int mBase = blockIdx.y * 128;
    const int nBase = blockIdx.x * 128;

    // --- staging coordinates (per thread: one 16-element segment) ---
    const int row = t >> 1;            // 0..127
    const int seg = t & 1;             // 0..1
    const int m   = mBase + row;
    const int b   = m / NTOK;
    const int hw  = m % NTOK;
    const int hh  = hw / NHH;
    const int ww  = hw % NHH;
    const float* xbase = x + (size_t)b * CCH * IMG * IMG
                           + (size_t)(hh * PATCH) * IMG + ww * PATCH;
    const unsigned short* btrow = Bt + (size_t)(nBase + row) * KDIM + seg * 16;
    const int sOff = row * 32 + seg * 16;            // element offset in tile

    // --- fragment coordinates ---
    const int half = lane >> 4;        // 0/1
    const int l15  = lane & 15;
    const int aOff = (wave * 16 + l15) * 32 + half * 8;

    v8f acc[8];
#pragma unroll
    for (int i = 0; i < 8; ++i) acc[i] = (v8f)(0.0f);

    float4 a0, a1, a2, a3;             // A prefetch registers (f32)
#if !USE_ASYNC_LDS
    uint4 bp0, bp1;                    // B prefetch registers (fallback)
#endif

    // ---- prologue: stage k-step 0 into buffer 0 ----
    {
        const float4* gp = (const float4*)(xbase + (size_t)seg * IMG); // c=0, p0=seg
        a0 = gp[0]; a1 = gp[1]; a2 = gp[2]; a3 = gp[3];
#if USE_ASYNC_LDS
        __builtin_amdgcn_global_load_async_to_lds_b128(to_glb(btrow), to_lds(&sB[0][sOff]), 0, 0);
        __builtin_amdgcn_global_load_async_to_lds_b128(to_glb(btrow), to_lds(&sB[0][sOff]), 16, 0);
#else
        const uint4* bg = (const uint4*)btrow;
        bp0 = bg[0]; bp1 = bg[1];
#endif
        uint4 u0, u1;
        u0.x = pk2bf(a0.x, a0.y); u0.y = pk2bf(a0.z, a0.w);
        u0.z = pk2bf(a1.x, a1.y); u0.w = pk2bf(a1.z, a1.w);
        u1.x = pk2bf(a2.x, a2.y); u1.y = pk2bf(a2.z, a2.w);
        u1.z = pk2bf(a3.x, a3.y); u1.w = pk2bf(a3.z, a3.w);
        uint4* da = (uint4*)&sA[0][sOff];
        da[0] = u0; da[1] = u1;
#if !USE_ASYNC_LDS
        uint4* db = (uint4*)&sB[0][sOff];
        db[0] = bp0; db[1] = bp1;
#endif
    }
#if USE_ASYNC_LDS
    __builtin_amdgcn_s_wait_asynccnt(0);
#endif
    __syncthreads();

    // ---- main loop: compute kk while staging kk+1 into the other buffer ----
    for (int kk = 0; kk < 96; ++kk) {
        const int buf  = kk & 1;
        const bool more = (kk < 95);

        if (more) {
            const int kn = kk + 1;
#if USE_ASYNC_LDS
            __builtin_amdgcn_global_load_async_to_lds_b128(
                to_glb(btrow + kn * 32), to_lds(&sB[buf ^ 1][sOff]), 0, 0);
            __builtin_amdgcn_global_load_async_to_lds_b128(
                to_glb(btrow + kn * 32), to_lds(&sB[buf ^ 1][sOff]), 16, 0);
#else
            const uint4* bg = (const uint4*)(btrow + kn * 32);
            bp0 = bg[0]; bp1 = bg[1];
#endif
            const int c  = kn >> 3;
            const int p0 = ((kn & 7) << 1) + seg;
            const float4* gp = (const float4*)(xbase + (size_t)c * IMG * IMG
                                                     + (size_t)p0 * IMG);
            a0 = gp[0]; a1 = gp[1]; a2 = gp[2]; a3 = gp[3];
        }

        // --- compute from buffer `buf` ---
        union Frag { v16bf v; uint4 q[2]; };
        Frag af, bfr[8];
        // A 16x32 bf16: lanes 0-15 -> K 0..7 (V0-3) & 16..23 (V4-7); lanes 16-31 +8
        const unsigned short* ap = &sA[buf][aOff];
        af.q[0] = *(const uint4*)(ap);
        af.q[1] = *(const uint4*)(ap + 16);
#pragma unroll
        for (int nt = 0; nt < 8; ++nt) {
            // B 32x16 bf16: lanes 0-15 -> K 0..7 & 16..23; lanes 16-31 +8
            const unsigned short* bp = &sB[buf][(nt * 16 + l15) * 32 + half * 8];
            bfr[nt].q[0] = *(const uint4*)(bp);
            bfr[nt].q[1] = *(const uint4*)(bp + 16);
        }
#pragma unroll
        for (int nt = 0; nt < 8; ++nt) {
            acc[nt] = __builtin_amdgcn_wmma_f32_16x16x32_bf16(
                false, af.v, false, bfr[nt].v, (short)0, acc[nt], false, false);
        }

        if (more) {
            uint4 u0, u1;
            u0.x = pk2bf(a0.x, a0.y); u0.y = pk2bf(a0.z, a0.w);
            u0.z = pk2bf(a1.x, a1.y); u0.w = pk2bf(a1.z, a1.w);
            u1.x = pk2bf(a2.x, a2.y); u1.y = pk2bf(a2.z, a2.w);
            u1.z = pk2bf(a3.x, a3.y); u1.w = pk2bf(a3.z, a3.w);
            uint4* da = (uint4*)&sA[buf ^ 1][sOff];
            da[0] = u0; da[1] = u1;
#if !USE_ASYNC_LDS
            uint4* db = (uint4*)&sB[buf ^ 1][sOff];
            db[0] = bp0; db[1] = bp1;
#endif
        }
#if USE_ASYNC_LDS
        __builtin_amdgcn_s_wait_asynccnt(0);
#endif
        __syncthreads();
    }

    // --- epilogue: D layout M = v + 8*half, N = l15 ---
#pragma unroll
    for (int v = 0; v < 8; ++v) {
        int r   = mBase + wave * 16 + v + 8 * half;
        int rb  = r / NTOK;
        int rhw = r % NTOK;
        float* orow = out + ((size_t)rb * 197 + 1 + rhw) * EMB;
        const float* pe = pos_embed + (size_t)(1 + rhw) * EMB;
#pragma unroll
        for (int nt = 0; nt < 8; ++nt) {
            int e = nBase + nt * 16 + l15;
            orow[e] = acc[nt][v] + proj_bias[e] + pe[e];
        }
    }
}

// ---------------------------------------------------------------------------
extern "C" void kernel_launch(void* const* d_in, const int* in_sizes, int n_in,
                              void* d_out, int out_size, void* d_ws, size_t ws_size,
                              hipStream_t stream) {
    const float* x   = (const float*)d_in[0];
    const float* wl  = (const float*)d_in[1];
    const float* w1  = (const float*)d_in[2];
    const float* b1  = (const float*)d_in[3];
    const float* w2  = (const float*)d_in[4];
    const float* b2  = (const float*)d_in[5];
    const float* pb  = (const float*)d_in[6];
    const float* pe  = (const float*)d_in[7];
    const float* cls = (const float*)d_in[8];
    float* out = (float*)d_out;

    float* hbuf = (float*)d_ws;                                   // 6 KB
    unsigned short* Bt = (unsigned short*)((char*)d_ws + 8192);   // 4.72 MB

    k_wlmlp  <<<1, 128, 0, stream>>>(wl, w1, b1, hbuf);
    k_kernels<<<W2N / 256, 256, 0, stream>>>(hbuf, w2, b2, Bt);
    k_cls    <<<(BATCH * EMB) / 256, 256, 0, stream>>>(cls, pe, out);
    k_gemm   <<<dim3(EMB / 128, 6272 / 128), 256, 0, stream>>>(x, Bt, pb, pe, out);
}
